// LesionToDiseaseMapper_21672404975767
// MI455X (gfx1250) — matrix operand order repaired and checked by reading, not compiled, not run on hardware
//
#include <hip/hip_runtime.h>
#include <hip/hip_bf16.h>

typedef __attribute__((ext_vector_type(16))) _Float16 v16h;
typedef __attribute__((ext_vector_type(8)))  _Float16 h8;
typedef __attribute__((ext_vector_type(4)))  _Float16 h4;
typedef __attribute__((ext_vector_type(8)))  float    v8f;

#define BB    8192
#define LL    32
#define CC    16
#define DD    256
#define HH    4
#define KVSTR 264            // padded half-stride: 132 dwords, bank-conflict free
#define NBLK  1024
#define ITERS 4              // NBLK * ITERS * 2 == BB

// ---------------- Prep kernel: one workgroup, builds P (f16), sbias, W ----------------
__global__ __launch_bounds__(256) void prep_kernel(
    const float* __restrict__ delta,   // (32,16)
    const float* __restrict__ dq,      // (16,256)
    const float* __restrict__ lnqg, const float* __restrict__ lnqb,
    const float* __restrict__ ipw,     // (768,256)
    const float* __restrict__ ipb,     // (768,)
    _Float16* __restrict__ Pout,       // (64,256) f16, scale folded
    float* __restrict__ SBout,         // (64,32)
    float* __restrict__ Wout)          // (32,16)
{
    __shared__ float qln[16 * 256];
    __shared__ float Qm[16 * 256];
    __shared__ float qbs[64];
    const int t = threadIdx.x;

    // LayerNorm of disease_query: 16 threads per row
    {
        const int row = t >> 4, sub = t & 15;
        float x[16]; float s = 0.f, ss = 0.f;
        #pragma unroll
        for (int i = 0; i < 16; i++) {
            x[i] = dq[row * 256 + sub * 16 + i];
            s += x[i]; ss += x[i] * x[i];
        }
        #pragma unroll
        for (int m = 1; m < 16; m <<= 1) { s += __shfl_xor(s, m, 32); ss += __shfl_xor(ss, m, 32); }
        const float mean = s * (1.f / 256.f);
        const float rstd = rsqrtf(ss * (1.f / 256.f) - mean * mean + 1e-5f);
        #pragma unroll
        for (int i = 0; i < 16; i++) {
            const int d = sub * 16 + i;
            qln[row * 256 + d] = (x[i] - mean) * rstd * lnqg[d] + lnqb[d];
        }
    }
    __syncthreads();

    // Q[c][e] = sum_d qln[c][d]*Wq[e][d] + bq[e]; thread owns column e = t
    {
        float acc[16];
        #pragma unroll
        for (int c = 0; c < 16; c++) acc[c] = 0.f;
        for (int d = 0; d < 256; d++) {
            const float wv = ipw[t * 256 + d];
            #pragma unroll
            for (int c = 0; c < 16; c++) acc[c] += qln[c * 256 + d] * wv;
        }
        #pragma unroll
        for (int c = 0; c < 16; c++) Qm[c * 256 + t] = acc[c] + ipb[t];
    }
    __syncthreads();

    // P[row=h*16+c][d] = scale * sum_k Q[c][h*64+k] * Wk[h*64+k][d]; thread owns d = t
    for (int row = 0; row < 64; row++) {
        const int h = row >> 4, c = row & 15;
        float a = 0.f;
        for (int k = 0; k < 64; k++)
            a += Qm[c * 256 + h * 64 + k] * ipw[(256 + h * 64 + k) * 256 + t];
        Pout[row * 256 + t] = (_Float16)(a * 0.125f);
    }
    // qb[row] = sum_k Q[c][h*64+k] * bk[h*64+k]
    if (t < 64) {
        const int h = t >> 4, c = t & 15;
        float a = 0.f;
        for (int k = 0; k < 64; k++) a += Qm[c * 256 + h * 64 + k] * ipb[256 + h * 64 + k];
        qbs[t] = a;
    }
    __syncthreads();

    // W = relu(prior + delta); sbias[row][l] = qb*scale + clip(2*W[l][c], -10, 10)
    for (int idx = t; idx < 512; idx += 256) {
        const int l = idx >> 4, c = idx & 15;
        const float m = ((l + c) % 3 == 0) ? 1.f : 0.f;
        Wout[idx] = fmaxf(m + delta[idx], 0.f);
    }
    for (int idx = t; idx < 2048; idx += 256) {
        const int rhc = idx >> 5, l = idx & 31;
        const int c = rhc & 15;
        const float m = ((l + c) % 3 == 0) ? 1.f : 0.f;
        const float Wv = fmaxf(m + delta[l * 16 + c], 0.f);
        const float bias = fminf(fmaxf(2.f * Wv, -10.f), 10.f);
        SBout[idx] = qbs[rhc] * 0.125f + bias;
    }
}

// ---------------- Main kernel ----------------
__global__ __launch_bounds__(256) void lesion_main_kernel(
    const float* __restrict__ le,     // (B,32,256)
    const float* __restrict__ probs,  // (B,32)
    const float* __restrict__ thr,    // (32,)
    const float* __restrict__ lng, const float* __restrict__ lnb,
    const _Float16* __restrict__ Pm,  // (64,256)
    const float* __restrict__ SB,     // (64,32)
    const float* __restrict__ WM,     // (32,16)
    float* __restrict__ out)          // (B,16)
{
    __shared__ alignas(16) _Float16 kvsh[2 * 32 * KVSTR];
    __shared__ float corr[32];

    const int tid  = threadIdx.x;
    const int wave = tid >> 5, lane = tid & 31;
    const int bs = wave >> 2;          // which batch of the pair this wave works on
    const int h  = wave & 3;           // head == M-tile
    const int lm = lane & 15;
    const int hi = lane >> 4;          // lane half

    // A fragments: constant across batches (16-bit A 16x32 layout)
    v16h afr[8];
    {
        const _Float16* base = Pm + (h * 16 + lm) * 256;
        #pragma unroll
        for (int ks = 0; ks < 8; ks++) {
            h8 lo = *(const h8*)(base + ks * 32 + (hi ? 16 : 0));
            h8 hh = *(const h8*)(base + ks * 32 + (hi ? 24 : 8));
            afr[ks] = __builtin_shufflevector(lo, hh, 0,1,2,3,4,5,6,7,8,9,10,11,12,13,14,15);
        }
    }
    // Bias and W fragments matching D layout (VGPR r: c=r lanes 0-15, c=8+r lanes 16-31)
    float sb0[8], sb1[8], wf0[8], wf1[8];
    #pragma unroll
    for (int r = 0; r < 8; r++) {
        const int rr = h * 16 + r + hi * 8;
        sb0[r] = SB[rr * 32 + lm];
        sb1[r] = SB[rr * 32 + 16 + lm];
        const int cc = r + hi * 8;
        wf0[r] = WM[lm * 16 + cc];
        wf1[r] = WM[(lm + 16) * 16 + cc];
    }
    if (tid < 32) corr[tid] = 0.f;

    const int rowl = tid >> 3, seg = tid & 7;   // LN: 8 threads per lesion row

    for (int it = 0; it < ITERS; it++) {
        const int pr = it * (int)gridDim.x + (int)blockIdx.x;
        const int b0 = pr * 2;

        // ---- LayerNorm both batches of the pair into LDS (f16) ----
        for (int s = 0; s < 2; s++) {
            const float* src = le + ((size_t)(b0 + s) * 32 + rowl) * 256;
            float4 x[8]; float sm = 0.f, sq = 0.f;
            #pragma unroll
            for (int i = 0; i < 8; i++) {
                x[i] = *(const float4*)(src + i * 32 + seg * 4);
                sm += x[i].x + x[i].y + x[i].z + x[i].w;
                sq += x[i].x * x[i].x + x[i].y * x[i].y + x[i].z * x[i].z + x[i].w * x[i].w;
            }
            #pragma unroll
            for (int m = 1; m < 8; m <<= 1) { sm += __shfl_xor(sm, m, 32); sq += __shfl_xor(sq, m, 32); }
            const float mean = sm * (1.f / 256.f);
            const float rstd = rsqrtf(sq * (1.f / 256.f) - mean * mean + 1e-5f);
            _Float16* dst = kvsh + s * (32 * KVSTR) + rowl * KVSTR;
            #pragma unroll
            for (int i = 0; i < 8; i++) {
                const int d = i * 32 + seg * 4;
                const float4 g = *(const float4*)(lng + d);
                const float4 b = *(const float4*)(lnb + d);
                h4 ph;
                ph[0] = (_Float16)((x[i].x - mean) * rstd * g.x + b.x);
                ph[1] = (_Float16)((x[i].y - mean) * rstd * g.y + b.y);
                ph[2] = (_Float16)((x[i].z - mean) * rstd * g.z + b.z);
                ph[3] = (_Float16)((x[i].w - mean) * rstd * g.w + b.w);
                *(h4*)(dst + d) = ph;
            }
        }
        __syncthreads();

        const int bb = b0 + bs;
        const float u0 = fmaxf(probs[bb * 32 + lm]      - thr[lm],      0.f);
        const float u1 = fmaxf(probs[bb * 32 + 16 + lm] - thr[16 + lm], 0.f);

        // ---- WMMA: scores = P(f16) x kv^T, M=16(h-tile) N=32 K=256 ----
        v8f a0 = {0.f,0.f,0.f,0.f,0.f,0.f,0.f,0.f};
        v8f a1 = {0.f,0.f,0.f,0.f,0.f,0.f,0.f,0.f};
        const _Float16* kb = kvsh + bs * (32 * KVSTR);
        #pragma unroll
        for (int ks = 0; ks < 8; ks++) {
            // B layout: lanes 0-15 hold K 0-15 (col n = lm), lanes 16-31 hold K 16-31
            const _Float16* q0 = kb + lm * KVSTR + ks * 32 + hi * 16;
            h8 b0lo = *(const h8*)q0;
            h8 b0hi = *(const h8*)(q0 + 8);
            v16h bf0 = __builtin_shufflevector(b0lo, b0hi, 0,1,2,3,4,5,6,7,8,9,10,11,12,13,14,15);
            a0 = __builtin_amdgcn_wmma_f32_16x16x32_f16(false, afr[ks], false, bf0,
                                                        (short)0, a0, false, false);
            const _Float16* q1 = kb + (16 + lm) * KVSTR + ks * 32 + hi * 16;
            h8 b1lo = *(const h8*)q1;
            h8 b1hi = *(const h8*)(q1 + 8);
            v16h bf1 = __builtin_shufflevector(b1lo, b1hi, 0,1,2,3,4,5,6,7,8,9,10,11,12,13,14,15);
            a1 = __builtin_amdgcn_wmma_f32_16x16x32_f16(false, afr[ks], false, bf1,
                                                        (short)0, a1, false, false);
        }

        // ---- softmax over l (32) + sum_l p*u*W per (h,c); butterfly in 16-lane halves ----
        #pragma unroll
        for (int r = 0; r < 8; r++) {
            float s0 = a0[r] + sb0[r];
            float s1 = a1[r] + sb1[r];
            float mx = fmaxf(s0, s1);
            #pragma unroll
            for (int m = 1; m < 16; m <<= 1) mx = fmaxf(mx, __shfl_xor(mx, m, 32));
            const float e0 = __expf(s0 - mx);
            const float e1 = __expf(s1 - mx);
            float den = e0 + e1;
            float num = e0 * u0 * wf0[r] + e1 * u1 * wf1[r];
            #pragma unroll
            for (int m = 1; m < 16; m <<= 1) { den += __shfl_xor(den, m, 32); num += __shfl_xor(num, m, 32); }
            const float val = num / den;
            if (lm == 0) atomicAdd(&corr[bs * 16 + r + hi * 8], val);  // ds_add_f32, sums heads
        }
        __syncthreads();

        if (tid < 32) {
            const int sb_ = tid >> 4, c = tid & 15;
            out[(size_t)(b0 + sb_) * 16 + c] = corr[tid] * 0.25f;   // mean over H=4 heads
            corr[tid] = 0.f;
        }
    }
}

extern "C" void kernel_launch(void* const* d_in, const int* in_sizes, int n_in,
                              void* d_out, int out_size, void* d_ws, size_t ws_size,
                              hipStream_t stream) {
    const float* le    = (const float*)d_in[0];
    const float* probs = (const float*)d_in[1];
    const float* thr   = (const float*)d_in[2];
    const float* delta = (const float*)d_in[3];
    const float* dq    = (const float*)d_in[4];
    const float* lnqg  = (const float*)d_in[5];
    const float* lnqb  = (const float*)d_in[6];
    const float* lnkg  = (const float*)d_in[7];
    const float* lnkb  = (const float*)d_in[8];
    const float* ipw   = (const float*)d_in[9];
    const float* ipb   = (const float*)d_in[10];
    float* out = (float*)d_out;

    _Float16* Pws = (_Float16*)d_ws;                       // 64*256*2  = 32768 B
    float*    SBw = (float*)((char*)d_ws + 32768);         // 64*32*4   =  8192 B
    float*    Wws = (float*)((char*)d_ws + 32768 + 8192);  // 32*16*4   =  2048 B

    prep_kernel<<<1, 256, 0, stream>>>(delta, dq, lnqg, lnqb, ipw, ipb, Pws, SBw, Wws);
    lesion_main_kernel<<<NBLK, 256, 0, stream>>>(le, probs, thr, lnkg, lnkb,
                                                 Pws, SBw, Wws, out);
}
